// VAE_1812476198978
// MI455X (gfx1250) — matrix operand order repaired and checked by reading, not compiled, not run on hardware
//
#include <hip/hip_runtime.h>
#include <hip/hip_bf16.h>
#include <math.h>

// ---------------------------------------------------------------------------
// Sizes (fixed by the reference)
// ---------------------------------------------------------------------------
#define NN   4096
#define TT   128
#define DIN  32
#define HH   256
#define GG   256
#define LATD 64
#define FUTD 128
#define LNEPS 1e-5f

typedef __attribute__((ext_vector_type(16))) __bf16 v16bf;
typedef __attribute__((ext_vector_type(8)))  float  v8f;

// ---------------------------------------------------------------------------
// WMMA fragment helpers (bf16 16x16x32, f32 accumulate)
//
// A (16x32 bf16) lane layout used here (same K order as B): lane l row m=l&15, half=l>>4,
//   elements 0..15 -> K = k0 + half*16 + 0..15   (one contiguous 32B run)
// B (32x16 bf16): pre-swizzled in workspace so each lane's 16 bf16 are
//   contiguous: block (kt,ct) of 32x16, lane l holds col=l&15,
//   K = kt*32 + (l>>4)*16 + 0..15  packed at blk*512 + l*16.
// C/D (16x16 f32): lane l col n=l&15, vgpr v row m = v + 8*(l>>4).
// ---------------------------------------------------------------------------
__device__ __forceinline__ v16bf frag_a(const __bf16* tile, int ldk, int m0,
                                        int kk, int lane) {
  int m = m0 + (lane & 15);
  int half = lane >> 4;
  const __bf16* p = tile + (size_t)m * ldk + kk * 32 + half * 16;
  union { v16bf v; uint4 q[2]; } u;
  u.q[0] = *reinterpret_cast<const uint4*>(p);
  u.q[1] = *reinterpret_cast<const uint4*>(p + 8);
  return u.v;
}

__device__ __forceinline__ v16bf frag_b(const __bf16* blkbase, int lane) {
  const __bf16* p = blkbase + lane * 16;
  union { v16bf v; uint4 q[2]; } u;
  u.q[0] = *reinterpret_cast<const uint4*>(p);
  u.q[1] = *reinterpret_cast<const uint4*>(p + 8);
  return u.v;
}

__device__ __forceinline__ v8f wmma_bf16(v16bf a, v16bf b, v8f c) {
  return __builtin_amdgcn_wmma_f32_16x16x32_bf16(
      false, a, false, b, (short)0, c, false, false);
}

__device__ __forceinline__ float sigf(float x) {
  return 1.0f / (1.0f + __expf(-x));
}

// ---------------------------------------------------------------------------
// Weight prep: fp32 [K,N] (or [N,K] with transpose) -> bf16 WMMA-B swizzle.
// ---------------------------------------------------------------------------
__global__ __launch_bounds__(256) void swizzle_w(const float* __restrict__ W,
                                                 __bf16* __restrict__ out,
                                                 int Ks, int Ns, int Kp, int Np,
                                                 int transpose) {
  int idx = blockIdx.x * 256 + threadIdx.x;
  int total = Kp * Np;
  if (idx >= total) return;
  int blk = idx >> 9;
  int within = idx & 511;
  int lane = within >> 4;
  int e = within & 15;
  int NT = Np >> 4;
  int kt = blk / NT;
  int ct = blk - kt * NT;
  int K = kt * 32 + (lane >> 4) * 16 + e;
  int col = ct * 16 + (lane & 15);
  float v = 0.0f;
  if (K < Ks && col < Ns)
    v = transpose ? W[(size_t)col * Ks + K] : W[(size_t)K * Ns + col];
  out[idx] = (__bf16)v;
}

__global__ __launch_bounds__(256) void combine_bias(
    const float* b0a, const float* b0b, const float* b1a, const float* b1b,
    const float* b2a, const float* b2b, float* out) {
  int i = blockIdx.x * 256 + threadIdx.x;
  if (i >= 3 * 1024) return;
  int l = i >> 10, j = i & 1023;
  const float* A = (l == 0) ? b0a : (l == 1) ? b1a : b2a;
  const float* B = (l == 0) ? b0b : (l == 1) ? b1b : b2b;
  out[i] = A[j] + B[j];
}

// ---------------------------------------------------------------------------
// Fused 3-layer LSTM (timestep-synchronous across layers) + LayerNorm.
// Block = 256 threads (8 waves), owns 32 nodes for all 128 timesteps.
// Wave w: row sub-tile mt=w&1 (16 rows), hidden-column group cg=w>>1 (64 cols).
// A fragments are re-read from LDS inside the K-loop (no register caching ->
// no scratch spills); B fragments are software double-buffered so the next
// global_load is in flight during the current WMMA.
// Layer loop is fully unrolled so the weight pointers stay as kernel args
// (address-space inference -> global_load_b128, not flat_load).
// ---------------------------------------------------------------------------
__global__ __launch_bounds__(256) void lstm_ln_kernel(
    const float* __restrict__ x,
    const __bf16* __restrict__ wih0, const __bf16* __restrict__ whh0,
    const __bf16* __restrict__ wih1, const __bf16* __restrict__ whh1,
    const __bf16* __restrict__ wih2, const __bf16* __restrict__ whh2,
    const float* __restrict__ bias3,
    const float* __restrict__ ln_g, const float* __restrict__ ln_b,
    float* __restrict__ hLN) {
  extern __shared__ char smem[];
  __bf16* hbuf = (__bf16*)smem;                          // 3*2*32*256 bf16
  float*  cbuf = (float*)(smem + 98304);                 // 3*32*256 f32
  __bf16* xbuf = (__bf16*)(smem + 98304 + 98304);        // 32*32 bf16
  float*  red  = (float*)(smem + 98304 + 98304 + 2048);  // 32*8*2 f32

  const int tid = threadIdx.x;
  const int lane = tid & 31;
  const int w = tid >> 5;
  const int mt = w & 1;        // 16-row sub-tile
  const int cg = w >> 1;       // hidden col group: cols [cg*64, cg*64+64)
  const int nodeBase = blockIdx.x * 32;

  for (int i = tid; i < 3 * 2 * 32 * 256; i += 256) hbuf[i] = (__bf16)0.0f;
  for (int i = tid; i < 3 * 32 * 256; i += 256) cbuf[i] = 0.0f;
  __syncthreads();

  for (int t = 0; t < TT; ++t) {
    const int cur = t & 1, nxt = cur ^ 1;
    // stage x_t for this node tile as bf16
    for (int i = tid; i < 32 * 32; i += 256) {
      int r = i >> 5, d = i & 31;
      xbuf[i] = (__bf16)x[(size_t)(nodeBase + r) * (TT * DIN) + t * DIN + d];
    }
    __syncthreads();

#pragma unroll
    for (int l = 0; l < 3; ++l) {
      // constant-folded after unroll: pointers stay global kernel args
      const __bf16* __restrict__ Wi = (l == 0) ? wih0 : (l == 1) ? wih1 : wih2;
      const __bf16* __restrict__ Wh = (l == 0) ? whh0 : (l == 1) ? whh1 : whh2;
      __bf16* hin  = hbuf + (size_t)(l * 2 + cur) * 32 * 256;
      __bf16* hout = hbuf + (size_t)(l * 2 + nxt) * 32 * 256;
      const __bf16* ainbuf =
          (l == 0) ? xbuf : hbuf + (size_t)((l - 1) * 2 + nxt) * 32 * 256;
      const int ain_ldk = (l == 0) ? 32 : 256;
      const int nk_in = (l == 0) ? 1 : 8;
      float* cL = cbuf + (size_t)l * 32 * 256;

      for (int jj = 0; jj < 4; ++jj) {
        const int j = cg * 4 + jj;  // hidden col tile (of 16)
        v8f gf[4];
#pragma unroll
        for (int g = 0; g < 4; ++g) {
          const int ct = g * 16 + j;  // of 64 gate col tiles
          float bv = bias3[l * 1024 + ct * 16 + (lane & 15)];
          v8f acc;
#pragma unroll
          for (int q = 0; q < 8; ++q) acc[q] = bv;

          // recurrent contribution: K = 256, B double-buffered
          v16bf b = frag_b(Wh + (size_t)ct * 512, lane);
#pragma unroll
          for (int kk = 0; kk < 8; ++kk) {
            const int kn = (kk < 7) ? kk + 1 : 7;
            v16bf bn = frag_b(Wh + (size_t)(kn * 64 + ct) * 512, lane);
            v16bf a = frag_a(hin, 256, mt * 16, kk, lane);
            acc = wmma_bf16(a, b, acc);
            b = bn;
          }
          // input contribution
          v16bf bi = frag_b(Wi + (size_t)ct * 512, lane);
#pragma unroll
          for (int kk = 0; kk < nk_in; ++kk) {
            const int kn = (kk < nk_in - 1) ? kk + 1 : nk_in - 1;
            v16bf bin = frag_b(Wi + (size_t)(kn * 64 + ct) * 512, lane);
            v16bf a = frag_a(ainbuf, ain_ldk, mt * 16, kk, lane);
            acc = wmma_bf16(a, bi, acc);
            bi = bin;
          }
          gf[g] = acc;
        }
        // LSTM cell on C-fragment layout
        const int half = lane >> 4;
        const int hcol = j * 16 + (lane & 15);
#pragma unroll
        for (int v = 0; v < 8; ++v) {
          const int r = mt * 16 + half * 8 + v;
          float i_ = sigf(gf[0][v]);
          float f_ = sigf(gf[1][v]);
          float g_ = tanhf(gf[2][v]);
          float o_ = sigf(gf[3][v]);
          float co = cL[r * 256 + hcol];
          float cn = f_ * co + i_ * g_;
          cL[r * 256 + hcol] = cn;
          hout[r * 256 + hcol] = (__bf16)(o_ * tanhf(cn));
        }
      }
      __syncthreads();
    }
  }

  // Final hidden of layer 2 is buffer ((TT-1)&1)^1 == 0. LayerNorm it.
  const __bf16* hF = hbuf + (size_t)(2 * 2 + 0) * 32 * 256;
  const int r = tid >> 3, seg = tid & 7;
  float s = 0.0f, ss = 0.0f;
  for (int cc = 0; cc < 32; ++cc) {
    float v = (float)hF[r * 256 + seg * 32 + cc];
    s += v; ss += v * v;
  }
  red[(r * 8 + seg) * 2] = s;
  red[(r * 8 + seg) * 2 + 1] = ss;
  __syncthreads();
  if (seg == 0) {
    float S = 0.0f, SS = 0.0f;
    for (int k = 0; k < 8; ++k) { S += red[(r * 8 + k) * 2]; SS += red[(r * 8 + k) * 2 + 1]; }
    float mean = S * (1.0f / 256.0f);
    float var = SS * (1.0f / 256.0f) - mean * mean;
    red[r * 16] = mean;
    red[r * 16 + 1] = rsqrtf(var + LNEPS);
  }
  __syncthreads();
  {
    float mean = red[r * 16], rstd = red[r * 16 + 1];
    for (int cc = 0; cc < 32; ++cc) {
      int col = seg * 32 + cc;
      float v = ((float)hF[r * 256 + col] - mean) * rstd * ln_g[col] + ln_b[col];
      hLN[(size_t)(nodeBase + r) * 256 + col] = v;
    }
  }
}

// ---------------------------------------------------------------------------
// Generic fused GEMM: C[rowTile, N] = act(A[rowTile, K] @ Bsw + bias)
// A fp32 (lda == K), B pre-swizzled bf16, 32 rows per block, 8 waves.
// act: 0 = none, 1 = relu. Writes to C[row*ldc + colOff + col].
// ---------------------------------------------------------------------------
__global__ __launch_bounds__(256) void gemm_bf16_kernel(
    const float* __restrict__ A, const __bf16* __restrict__ Bsw,
    const float* __restrict__ bias, float* __restrict__ C,
    int N, int Nreal, int K, int ldc, int colOff, int act) {
  extern __shared__ char smem[];
  __bf16* As = (__bf16*)smem;  // [32][K]
  const int tid = threadIdx.x;
  const int lane = tid & 31;
  const int w = tid >> 5;
  const int mt = w & 1;
  const int cgrp = w >> 1;
  const int rowBase = blockIdx.x * 32;

  const int kshift = __builtin_ctz((unsigned)K);  // K is a power of two
  for (int i = tid; i < 32 * K; i += 256) {
    int r = i >> kshift, cc = i & (K - 1);
    As[i] = (__bf16)A[(size_t)(rowBase + r) * K + cc];
  }
  __syncthreads();

  const int NT = N >> 4, KT = K >> 5;
  for (int ct = cgrp; ct < NT; ct += 4) {
    const int col = ct * 16 + (lane & 15);
    float bv = (bias != nullptr && col < Nreal) ? bias[col] : 0.0f;
    v8f acc;
#pragma unroll
    for (int q = 0; q < 8; ++q) acc[q] = bv;
    v16bf b = frag_b(Bsw + (size_t)ct * 512, lane);
    for (int kk = 0; kk < KT; ++kk) {
      const int kn = (kk + 1 < KT) ? kk + 1 : kk;
      v16bf bn = frag_b(Bsw + (size_t)(kn * NT + ct) * 512, lane);
      v16bf a = frag_a(As, K, mt * 16, kk, lane);
      acc = wmma_bf16(a, b, acc);
      b = bn;
    }
    if (act == 1) {
#pragma unroll
      for (int q = 0; q < 8; ++q) acc[q] = fmaxf(acc[q], 0.0f);
    }
    const int half = lane >> 4;
    const int colAbs = colOff + col;
#pragma unroll
    for (int v = 0; v < 8; ++v)
      C[(size_t)(rowBase + mt * 16 + half * 8 + v) * ldc + colAbs] = acc[v];
  }
}

// ---------------------------------------------------------------------------
// GCN support kernels
// ---------------------------------------------------------------------------
__global__ __launch_bounds__(256) void deg_init(float* deg, int n) {
  int i = blockIdx.x * 256 + threadIdx.x;
  if (i < n) deg[i] = 1.0f;  // self loop
}
__global__ __launch_bounds__(256) void deg_accum(const int* __restrict__ ei,
                                                 float* deg, int E) {
  int e = blockIdx.x * 256 + threadIdx.x;
  if (e < E) unsafeAtomicAdd(&deg[ei[E + e]], 1.0f);
}
__global__ __launch_bounds__(256) void deg_inv(const float* deg, float* dinv, int n) {
  int i = blockIdx.x * 256 + threadIdx.x;
  if (i < n) dinv[i] = rsqrtf(deg[i]);
}
__global__ __launch_bounds__(256) void gcn_self(const float* __restrict__ xw,
                                                const float* __restrict__ dinv,
                                                const float* __restrict__ b,
                                                float* __restrict__ out) {
  int i = blockIdx.x * 256 + threadIdx.x;
  if (i >= NN * GG) return;
  int n = i >> 8, c = i & 255;
  float di = dinv[n];
  out[i] = b[c] + di * di * xw[i];
}
__global__ __launch_bounds__(256) void gcn_edges(const int* __restrict__ ei,
                                                 const float* __restrict__ xw,
                                                 const float* __restrict__ dinv,
                                                 float* __restrict__ out, int E) {
  int gid = blockIdx.x * 256 + threadIdx.x;
  int e = gid >> 5, lane = gid & 31;
  if (e >= E) return;
  int s = ei[e], d = ei[E + e];
  float nrm = dinv[s] * dinv[d];
  const float* src = xw + (size_t)s * GG;
  float* dst = out + (size_t)d * GG;
#pragma unroll
  for (int u = 0; u < 8; ++u) {
    int c = lane * 8 + u;
    unsafeAtomicAdd(&dst[c], src[c] * nrm);
  }
}

// ---------------------------------------------------------------------------
// Mixture reparameterization + head transforms.
// Deterministic counter-based hash RNG (fixed "seed 42" analogue).
// ---------------------------------------------------------------------------
__device__ __forceinline__ float urand(unsigned a, unsigned b) {
  unsigned x = a * 0x9E3779B1u ^ (b + 42u) * 0x85EBCA77u;
  x ^= x >> 16; x *= 0x7FEB352Du;
  x ^= x >> 15; x *= 0x846CA68Bu;
  x ^= x >> 16;
  return ((float)(x >> 8) + 0.5f) * (1.0f / 16777216.0f);
}

__global__ __launch_bounds__(256) void sample_kernel(
    const float* __restrict__ zraw, const float* __restrict__ pi,
    float* __restrict__ z, float* __restrict__ out) {
  int idx = blockIdx.x * 256 + threadIdx.x;
  if (idx >= NN * LATD) return;
  int n = idx >> 6, j = idx & 63;
  const float* r = zraw + (size_t)n * 320;
  float zm = r[j];
  float zl = r[64 + j];
  float zsc = __expf(r[128 + j]);
  float zsh = fmaxf(r[192 + j], 1e-6f);
  float zg = r[256 + j];
  // tuple outputs (after rec which occupies [0, NN*FUTD))
  out[NN * FUTD + 0 * NN * LATD + idx] = zm;
  out[NN * FUTD + 1 * NN * LATD + idx] = zl;
  out[NN * FUTD + 2 * NN * LATD + idx] = zsc;
  out[NN * FUTD + 3 * NN * LATD + idx] = zsh;
  out[NN * FUTD + 4 * NN * LATD + idx] = zg;

  float u1 = urand(idx, 1u), u2 = urand(idx, 2u);
  float eps = sqrtf(fmaxf(-2.0f * __logf(fmaxf(u1, 1e-12f)), 0.0f)) *
              __cosf(6.2831853f * u2);
  float zgauss = zm + __expf(0.5f * zl) * eps;
  float ug = fminf(urand(idx, 3u), 0.999999f);
  float zgpd = zsc / zsh * (__expf(-zsh * log1pf(-ug)) - 1.0f);
  float ub = urand(idx, 4u);
  float zbern = (ub < sigf(zg)) ? 1.0f : 0.0f;
  float e0 = __expf(pi[0]), e1 = __expf(pi[1]), e2 = __expf(pi[2]);
  float inv = 1.0f / (e0 + e1 + e2);
  float p0 = e0 * inv, p1 = e1 * inv;
  float ch = urand((unsigned)n, 5u);
  z[idx] = ch < p0 ? zgauss : (ch < p0 + p1 ? zgpd : zbern);
}

// ---------------------------------------------------------------------------
// Host launcher
// ---------------------------------------------------------------------------
extern "C" void kernel_launch(void* const* d_in, const int* in_sizes, int n_in,
                              void* d_out, int out_size, void* d_ws, size_t ws_size,
                              hipStream_t stream) {
  (void)n_in; (void)out_size; (void)ws_size;
  const float* x   = (const float*)d_in[0];
  const int*   ei  = (const int*)d_in[1];
  const int E = in_sizes[1] / 2;
  const float* Wih0 = (const float*)d_in[2];
  const float* Whh0 = (const float*)d_in[3];
  const float* bih0 = (const float*)d_in[4];
  const float* bhh0 = (const float*)d_in[5];
  const float* Wih1 = (const float*)d_in[6];
  const float* Whh1 = (const float*)d_in[7];
  const float* bih1 = (const float*)d_in[8];
  const float* bhh1 = (const float*)d_in[9];
  const float* Wih2 = (const float*)d_in[10];
  const float* Whh2 = (const float*)d_in[11];
  const float* bih2 = (const float*)d_in[12];
  const float* bhh2 = (const float*)d_in[13];
  const float* ln_g = (const float*)d_in[14];
  const float* ln_b = (const float*)d_in[15];
  const float* gcnW = (const float*)d_in[16];
  const float* gcnb = (const float*)d_in[17];
  const float* fc1W = (const float*)d_in[18];
  const float* fc1b = (const float*)d_in[19];
  const float* fc2W = (const float*)d_in[20];
  const float* fc2b = (const float*)d_in[21];
  const float* headW[5] = {(const float*)d_in[22], (const float*)d_in[24],
                           (const float*)d_in[26], (const float*)d_in[28],
                           (const float*)d_in[30]};
  const float* headB[5] = {(const float*)d_in[23], (const float*)d_in[25],
                           (const float*)d_in[27], (const float*)d_in[29],
                           (const float*)d_in[31]};
  const float* d1W = (const float*)d_in[32];
  const float* d1b = (const float*)d_in[33];
  const float* d2W = (const float*)d_in[34];
  const float* d2b = (const float*)d_in[35];
  const float* doW = (const float*)d_in[36];
  const float* dob = (const float*)d_in[37];
  const float* pi  = (const float*)d_in[38];
  float* out = (float*)d_out;

  // workspace carve-out
  char* ws = (char*)d_ws;
  size_t off = 0;
  auto carve = [&](size_t bytes) {
    char* p = ws + off;
    off += (bytes + 255) & ~(size_t)255;
    return p;
  };
  __bf16* wih0sw = (__bf16*)carve(32 * 1024 * 2);
  __bf16* whh0sw = (__bf16*)carve(256 * 1024 * 2);
  __bf16* wih1sw = (__bf16*)carve(256 * 1024 * 2);
  __bf16* whh1sw = (__bf16*)carve(256 * 1024 * 2);
  __bf16* wih2sw = (__bf16*)carve(256 * 1024 * 2);
  __bf16* whh2sw = (__bf16*)carve(256 * 1024 * 2);
  float*  bias3  = (float*)carve(3 * 1024 * 4);
  __bf16* gcnWsw = (__bf16*)carve(256 * 256 * 2);
  __bf16* fc1sw  = (__bf16*)carve(256 * 128 * 2);
  __bf16* fc2sw  = (__bf16*)carve(128 * 64 * 2);
  __bf16* headsw = (__bf16*)carve(5 * 64 * 64 * 2);
  __bf16* d1sw   = (__bf16*)carve(64 * 128 * 2);
  __bf16* d2sw   = (__bf16*)carve(128 * 512 * 2);
  __bf16* dosw   = (__bf16*)carve(512 * 128 * 2);
  float* hLN    = (float*)carve((size_t)NN * 256 * 4);
  float* xw     = (float*)carve((size_t)NN * 256 * 4);
  float* gcnOut = (float*)carve((size_t)NN * 256 * 4);
  float* deg    = (float*)carve((size_t)NN * 4);
  float* dinv   = (float*)carve((size_t)NN * 4);
  float* hid1   = (float*)carve((size_t)NN * 128 * 4);
  float* hid2   = (float*)carve((size_t)NN * 64 * 4);
  float* zraw   = (float*)carve((size_t)NN * 320 * 4);
  float* zbuf   = (float*)carve((size_t)NN * 64 * 4);
  float* dec1   = (float*)carve((size_t)NN * 128 * 4);
  float* dec2   = (float*)carve((size_t)NN * 512 * 4);

  auto gridFor = [](int total) { return dim3((total + 255) / 256); };

  // ---- weight prep (transpose + bf16 + WMMA-B swizzle) ----
  swizzle_w<<<gridFor(32 * 1024), 256, 0, stream>>>(Wih0, wih0sw, 32, 1024, 32, 1024, 1);
  swizzle_w<<<gridFor(256 * 1024), 256, 0, stream>>>(Whh0, whh0sw, 256, 1024, 256, 1024, 1);
  swizzle_w<<<gridFor(256 * 1024), 256, 0, stream>>>(Wih1, wih1sw, 256, 1024, 256, 1024, 1);
  swizzle_w<<<gridFor(256 * 1024), 256, 0, stream>>>(Whh1, whh1sw, 256, 1024, 256, 1024, 1);
  swizzle_w<<<gridFor(256 * 1024), 256, 0, stream>>>(Wih2, wih2sw, 256, 1024, 256, 1024, 1);
  swizzle_w<<<gridFor(256 * 1024), 256, 0, stream>>>(Whh2, whh2sw, 256, 1024, 256, 1024, 1);
  combine_bias<<<gridFor(3 * 1024), 256, 0, stream>>>(bih0, bhh0, bih1, bhh1, bih2, bhh2, bias3);
  swizzle_w<<<gridFor(256 * 256), 256, 0, stream>>>(gcnW, gcnWsw, 256, 256, 256, 256, 0);
  swizzle_w<<<gridFor(256 * 128), 256, 0, stream>>>(fc1W, fc1sw, 256, 128, 256, 128, 0);
  swizzle_w<<<gridFor(128 * 64), 256, 0, stream>>>(fc2W, fc2sw, 128, 64, 128, 64, 0);
  for (int h = 0; h < 5; ++h)
    swizzle_w<<<gridFor(64 * 64), 256, 0, stream>>>(headW[h], headsw + h * 64 * 64, 64, 64, 64, 64, 0);
  swizzle_w<<<gridFor(64 * 128), 256, 0, stream>>>(d1W, d1sw, 64, 128, 64, 128, 0);
  swizzle_w<<<gridFor(128 * 512), 256, 0, stream>>>(d2W, d2sw, 128, 500, 128, 512, 0);
  swizzle_w<<<gridFor(512 * 128), 256, 0, stream>>>(doW, dosw, 500, 128, 512, 128, 0);

  // ---- fused 3-layer LSTM + LayerNorm ----
  {
    size_t lstmSmem = 98304 /*h*/ + 98304 /*c*/ + 2048 /*x*/ + 2048 /*red*/;
    lstm_ln_kernel<<<dim3(NN / 32), 256, lstmSmem, stream>>>(
        x, wih0sw, whh0sw, wih1sw, whh1sw, wih2sw, whh2sw, bias3, ln_g, ln_b, hLN);
  }

  const dim3 gemmGrid(NN / 32);
  auto gemm = [&](const float* A, const __bf16* B, const float* b, float* C,
                  int N, int Nreal, int K, int ldc, int colOff, int act) {
    gemm_bf16_kernel<<<gemmGrid, 256, (size_t)32 * K * 2, stream>>>(
        A, B, b, C, N, Nreal, K, ldc, colOff, act);
  };

  // ---- GCN: xw = hLN @ W ; symmetric-norm aggregate + bias ----
  gemm(hLN, gcnWsw, nullptr, xw, 256, 256, 256, 256, 0, 0);
  deg_init<<<gridFor(NN), 256, 0, stream>>>(deg, NN);
  deg_accum<<<gridFor(E), 256, 0, stream>>>(ei, deg, E);
  deg_inv<<<gridFor(NN), 256, 0, stream>>>(deg, dinv, NN);
  gcn_self<<<gridFor(NN * GG), 256, 0, stream>>>(xw, dinv, gcnb, gcnOut);
  gcn_edges<<<gridFor(E * 32), 256, 0, stream>>>(ei, xw, dinv, gcnOut, E);

  // ---- encoder MLP + heads ----
  gemm(gcnOut, fc1sw, fc1b, hid1, 128, 128, 256, 128, 0, 1);
  gemm(hid1, fc2sw, fc2b, hid2, 64, 64, 128, 64, 0, 1);
  for (int h = 0; h < 5; ++h)
    gemm(hid2, headsw + h * 64 * 64, headB[h], zraw, 64, 64, 64, 320, h * 64, 0);

  // ---- mixture sampling + tuple outputs ----
  sample_kernel<<<gridFor(NN * LATD), 256, 0, stream>>>(zraw, pi, zbuf, out);

  // ---- decoder ----
  gemm(zbuf, d1sw, d1b, dec1, 128, 128, 64, 128, 0, 1);
  gemm(dec1, d2sw, d2b, dec2, 512, 500, 128, 512, 0, 1);
  gemm(dec2, dosw, dob, out /*rec at offset 0*/, 128, 128, 512, 128, 0, 0);
}
